// Attention_10179072491905
// MI455X (gfx1250) — matrix operand order repaired and checked by reading, not compiled, not run on hardware
//
#include <hip/hip_runtime.h>
#include <hip/hip_bf16.h>

// ---- CDNA5 WMMA vector types ----
typedef __attribute__((ext_vector_type(16))) __bf16 v16bf;
typedef __attribute__((ext_vector_type(8)))  float  v8f;

#define S_LEN  4096
#define NHEAD  16
#define DHEAD  64
#define DMODEL (NHEAD * DHEAD)
#define BM     128   // query rows per workgroup (8 waves x 16 rows)
#define BN     64    // keys per tile

__global__ __launch_bounds__(256)
void fa_fwd_causal_bf16_wmma(const float* __restrict__ Qg,
                             const float* __restrict__ Kg,
                             const float* __restrict__ Vg,
                             float* __restrict__ Og)
{
    // Double-buffered K/V tiles (ping-pong) + per-wave P relayout scratch.
    __shared__ __bf16 sK[2][DHEAD * BN];   // K tile, transposed: [d][n]
    __shared__ __bf16 sV[2][BN * DHEAD];   // V tile, row-major:  [n][d]
    __shared__ __bf16 sP[8 * 16 * BN];     // per-wave 16x64 P (C->A relayout)

    const int tid  = threadIdx.x;
    const int lane = tid & 31;
    const int wave = tid >> 5;
    const int ln   = lane & 15;      // N / M position within half-wave
    const int hi   = lane >> 4;      // half-wave select

    const int qb  = blockIdx.x;
    const int h   = blockIdx.y;
    const int q0  = qb * BM;
    const int q0w = q0 + wave * 16;  // this wave's first query row

    // Per-thread staging coordinates: 4 float4 chunks per tensor per tile.
    const int st_n = tid >> 4;              // base n for i=0 chunk
    const int st_d = (tid & 15) << 2;       // d offset (float4 aligned)

    // ---- Q fragments: A-layout 16x32 bf16, two K-chunks for Dh=64 ----
    // A layout (16-bit, 16x32): lane(0..15)=M, hi-half shifts K by 8;
    // VGPR v, half j -> K = (v<4?0:16) + hi*8 + 2*(v&3) + j   (j-pairs contiguous)
    v16bf aq[2];
#pragma unroll
    for (int kc = 0; kc < 2; ++kc) {
#pragma unroll
        for (int v = 0; v < 8; ++v) {
            const int kk = ((v < 4) ? 0 : 16) + hi * 8 + ((v & 3) << 1);
            const float2 f = *(const float2*)
                &Qg[(size_t)(q0w + ln) * DMODEL + h * DHEAD + kc * 32 + kk];
            aq[kc][2 * v + 0] = (__bf16)f.x;
            aq[kc][2 * v + 1] = (__bf16)f.y;
        }
    }

    // all-ones B fragment: row-sum reducer for the matrix pipe
    v16bf ones;
#pragma unroll
    for (int e = 0; e < 16; ++e) ones[e] = (__bf16)1.0f;

    const v8f vzero = {};
    v8f acc_o[4] = {vzero, vzero, vzero, vzero};   // O accum, 4 d-tiles of 16
    float m_r[8], l_r[8];                          // per-row running max / sum
#pragma unroll
    for (int r = 0; r < 8; ++r) { m_r[r] = -1e30f; l_r[r] = 0.0f; }

    const int   nTiles = q0 / BN + 2;   // causal: last tile overlaps diagonal
    const float scale  = 0.125f;        // 1/sqrt(64)

    float4 kr[4], vr[4];                // in-flight next-tile registers

    auto load_tile = [&](int n0) {
#pragma unroll
        for (int i = 0; i < 4; ++i) {
            const int n = st_n + i * 16;
            const size_t g = (size_t)(n0 + n) * DMODEL + h * DHEAD + st_d;
            kr[i] = *(const float4*)&Kg[g];
            vr[i] = *(const float4*)&Vg[g];
        }
    };

    load_tile(0);                       // prime the pipeline

    for (int it = 0; it < nTiles; ++it) {
        const int n0  = it * BN;
        const int cur = it & 1;

        // ---- drain in-flight registers into LDS buf[cur] (fp32 -> bf16) ----
        __bf16* skc = sK[cur];
        __bf16* svc = sV[cur];
#pragma unroll
        for (int i = 0; i < 4; ++i) {
            const int n = st_n + i * 16;
            skc[(st_d + 0) * BN + n] = (__bf16)kr[i].x;   // transposed scatter
            skc[(st_d + 1) * BN + n] = (__bf16)kr[i].y;
            skc[(st_d + 2) * BN + n] = (__bf16)kr[i].z;
            skc[(st_d + 3) * BN + n] = (__bf16)kr[i].w;
            __bf16* vp = &svc[n * DHEAD + st_d];          // contiguous b64 store
            vp[0] = (__bf16)vr[i].x; vp[1] = (__bf16)vr[i].y;
            vp[2] = (__bf16)vr[i].z; vp[3] = (__bf16)vr[i].w;
        }

        // ---- issue next tile's global loads; they stay in flight across
        //      the barrier and the whole compute phase ----
        if (it + 1 < nTiles) {
            load_tile(n0 + BN);
            if (it + 2 < nTiles) {      // gfx1250 global_prefetch_b8, 2 ahead
                const size_t gp = (size_t)(n0 + 2 * BN + st_n) * DMODEL
                                  + h * DHEAD + st_d;
                __builtin_prefetch(&Kg[gp], 0, 1);
                __builtin_prefetch(&Vg[gp], 0, 1);
            }
        }

        // Single barrier per tile: orders stage(it)->compute(it) and, via the
        // previous iteration's barrier, compute(it-2)->overwrite of buf[cur].
        __syncthreads();

        // ---- S = Q * K^T : 16x64 per wave, 8 WMMAs ----
        v8f sc[4] = {vzero, vzero, vzero, vzero};
#pragma unroll
        for (int kc = 0; kc < 2; ++kc) {
#pragma unroll
            for (int t = 0; t < 4; ++t) {
                v16bf b;   // B-layout: lane = N(n), VGPR v half j -> K(d) as in A layout
#pragma unroll
                for (int v = 0; v < 8; ++v) {
#pragma unroll
                    for (int j = 0; j < 2; ++j)
                        b[2 * v + j] =
                            skc[(kc * 32 + ((v < 4) ? 0 : 16) + hi * 8 + ((v & 3) << 1) + j) * BN + t * 16 + ln];
                }
                sc[t] = __builtin_amdgcn_wmma_f32_16x16x32_bf16(
                    false, aq[kc], false, b, (short)0, sc[t], false, false);
            }
        }

        // ---- scale + causal mask + online row max (C-layout registers) ----
        float corr[8], mnew[8];
#pragma unroll
        for (int r = 0; r < 8; ++r) {
            const int qrow = q0w + r + 8 * hi;     // C layout: M = r + 8*hi
            float x = -1e30f;
#pragma unroll
            for (int t = 0; t < 4; ++t) {
                float s = sc[t][r] * scale;
                if (n0 + t * 16 + ln > qrow) s = -1e30f;   // causal
                sc[t][r] = s;
                x = fmaxf(x, s);
            }
#pragma unroll
            for (int off = 8; off >= 1; off >>= 1)
                x = fmaxf(x, __shfl_xor(x, off, 16));      // row max over N
            const float mn = fmaxf(m_r[r], x);
            mnew[r] = mn;
            corr[r] = __expf(m_r[r] - mn);
            m_r[r]  = mn;
        }

        // ---- P = exp(s - m), spill to LDS (C -> A relayout) ----
        __bf16* pw = &sP[wave * (16 * BN)];
#pragma unroll
        for (int r = 0; r < 8; ++r) {
#pragma unroll
            for (int t = 0; t < 4; ++t) {
                const float p = __expf(sc[t][r] - mnew[r]);
                pw[(r + 8 * hi) * BN + t * 16 + ln] = (__bf16)p;
            }
        }

        // ---- reload P as A fragments (shared by row-sum and PV) ----
        v16bf ap[2];
#pragma unroll
        for (int kc = 0; kc < 2; ++kc) {
#pragma unroll
            for (int v = 0; v < 8; ++v) {
#pragma unroll
                for (int j = 0; j < 2; ++j) {
                    const int kk = ((v < 4) ? 0 : 16) + hi * 8 + ((v & 3) << 1) + j;
                    ap[kc][2 * v + j] = pw[ln * BN + kc * 32 + kk];
                }
            }
        }

        // ---- row sums on the matrix pipe: rs[m][*] = P x ones ----
        // D[m][n] = sum_k P[m,k]; lane's VGPR r = rowsum of row (r + 8*hi),
        // exactly matching the per-half-wave stats convention.
        v8f rs = vzero;
        rs = __builtin_amdgcn_wmma_f32_16x16x32_bf16(
            false, ap[0], false, ones, (short)0, rs, false, false);
        rs = __builtin_amdgcn_wmma_f32_16x16x32_bf16(
            false, ap[1], false, ones, (short)0, rs, false, false);

        // ---- online update: l, O rescale ----
#pragma unroll
        for (int r = 0; r < 8; ++r) {
            l_r[r] = l_r[r] * corr[r] + rs[r];
#pragma unroll
            for (int t = 0; t < 4; ++t)
                acc_o[t][r] *= corr[r];
        }

        // ---- O += P * V : 8 WMMAs ----
#pragma unroll
        for (int kc = 0; kc < 2; ++kc) {
#pragma unroll
            for (int t = 0; t < 4; ++t) {
                v16bf bv;  // lane = N(d), VGPR v half j -> K(n) as in A layout
#pragma unroll
                for (int v = 0; v < 8; ++v) {
#pragma unroll
                    for (int j = 0; j < 2; ++j)
                        bv[2 * v + j] =
                            svc[(kc * 32 + ((v < 4) ? 0 : 16) + hi * 8 + ((v & 3) << 1) + j) * DHEAD + t * 16 + ln];
                }
                acc_o[t] = __builtin_amdgcn_wmma_f32_16x16x32_bf16(
                    false, ap[kc], false, bv, (short)0, acc_o[t], false, false);
            }
        }
        // no trailing barrier: ping-pong buffer + next iteration's barrier
        // provide the ordering for buf reuse.
    }

    // ---- epilogue: O / l, fp32 out (coalesced across lanes via ln) ----
#pragma unroll
    for (int r = 0; r < 8; ++r) {
        const float inv = 1.0f / l_r[r];
#pragma unroll
        for (int t = 0; t < 4; ++t) {
            Og[(size_t)(q0w + r + 8 * hi) * DMODEL + h * DHEAD + t * 16 + ln] =
                acc_o[t][r] * inv;
        }
    }
}

extern "C" void kernel_launch(void* const* d_in, const int* in_sizes, int n_in,
                              void* d_out, int out_size, void* d_ws, size_t ws_size,
                              hipStream_t stream) {
    // setup_inputs order: query, key, value, mask, num_head, dim_head
    const float* Q = (const float*)d_in[0];
    const float* K = (const float*)d_in[1];
    const float* V = (const float*)d_in[2];
    // mask (d_in[3]) is the causal mask -> applied analytically in-kernel.
    float* O = (float*)d_out;

    dim3 grid(S_LEN / BM, NHEAD);   // 32 x 16 = 512 workgroups
    fa_fwd_causal_bf16_wmma<<<grid, 256, 0, stream>>>(Q, K, V, O);
}
